// CrystalGraphConvNet_27573690040696
// MI455X (gfx1250) — matrix operand order repaired and checked by reading, not compiled, not run on hardware
//
#include <hip/hip_runtime.h>
#include <stdint.h>
#include <stddef.h>

// ---------------------------------------------------------------------------
// CrystalGraphConvNet on MI455X (gfx1250, wave32, WMMA).
// Dominant cost: 3x GEMM [262144 x 320] @ [320 x 256] (~129 GFLOP) -> bf16
// WMMA with f32 accumulation. W K-slabs staged in LDS via
// GLOBAL_LOAD_ASYNC_TO_LDS_B128 so each block reads W exactly once: 4x less
// L2 traffic than per-wave streaming.
// Workspace layout (<= ~51 MB):
//   [0,8MB)      nf0  f32  [B*N,128]
//   [8MB,16MB)   nf1  f32  [B*N,128]
//   [16MB,48MB)  ef   bf16 [B*N*M,64]
//   [48MB,...)   Wbf1/2/3 bf16 [256,320] each, DA bf16 [256,256],
//                nffT bf16 [B,64,256] (transposed final features)
// ---------------------------------------------------------------------------

typedef __bf16 bf16_t;
typedef __attribute__((ext_vector_type(16))) __bf16 v16bf;
typedef __attribute__((ext_vector_type(8)))  float  v8f;

#define BB   64
#define NN   256
#define MM   16
#define ONF_ 92
#define OEF_ 41
#define NFD  128
#define EFD  64
#define KTOT 320   // 2*NFD + EFD
#define KPAD 328   // padded LDS row (bank-conflict-free ds_load_b128)
#define WSP  40    // W slab padded row: 32 data + 8 pad bf16
#define FH   64    // FINAL/2

#if defined(__has_builtin)
#if __has_builtin(__builtin_amdgcn_global_load_async_to_lds_b128)
#define ASYNC_LDS 1
#endif
#endif
#ifndef ASYNC_LDS
#define ASYNC_LDS 0
#endif

// Exact parameter types per hipcc diagnostic: v4i (vector_size 16) pointers.
typedef int v4i_ __attribute__((vector_size(16)));
typedef __attribute__((address_space(1))) v4i_* g4p;
typedef __attribute__((address_space(3))) v4i_* l4p;

static __device__ __forceinline__ unsigned short f2bf_bits(float f) {
  union { float f; unsigned u; } v; v.f = f;
  unsigned r = (v.u + 0x7FFFu + ((v.u >> 16) & 1u)) >> 16;  // round-nearest-even
  return (unsigned short)r;
}
static __device__ __forceinline__ bf16_t f2bf(float f) {
  union { unsigned short s; bf16_t b; } o; o.s = f2bf_bits(f); return o.b;
}
// Load two 8-element (16B) bf16 runs into one 16-wide fragment register set.
static __device__ __forceinline__ v16bf ld16bf(const bf16_t* p0, const bf16_t* p1) {
  union { v16bf v; uint4 q[2]; } u;
  u.q[0] = *(const uint4*)p0;
  u.q[1] = *(const uint4*)p1;
  return u.v;
}
static __device__ __forceinline__ float sigmoidf_(float x) {
  return 1.0f / (1.0f + __expf(-x));
}
static __device__ __forceinline__ float softplusf_(float x) {
  return fmaxf(x, 0.0f) + log1pf(__expf(-fabsf(x)));
}

// --------------------------- small prep kernels ----------------------------

__global__ __launch_bounds__(256) void k_cvt_bf16(const float* __restrict__ src,
                                                  bf16_t* __restrict__ dst, int n) {
  int i = blockIdx.x * 256 + threadIdx.x;
  if (i < n) dst[i] = f2bf(src[i]);
}

__global__ __launch_bounds__(256) void k_da(const float* __restrict__ dis,
                                            const float* __restrict__ w,
                                            const float* __restrict__ b,
                                            bf16_t* __restrict__ da) {
  int i = blockIdx.x * 256 + threadIdx.x;     // 256*256 launched exactly
  da[i] = f2bf(sigmoidf_(w[0] * dis[i] + b[0]));
}

// nf0[r,f] = node_fea[r,:92] . Wn[f,:92] + bn[f]
__global__ __launch_bounds__(128) void k_embed_node(const float* __restrict__ x,
                                                    const float* __restrict__ Wn,
                                                    const float* __restrict__ bn,
                                                    float* __restrict__ nf) {
  __shared__ float row[ONF_];
  int r = blockIdx.x;
  for (int t = threadIdx.x; t < ONF_; t += 128) row[t] = x[(size_t)r * ONF_ + t];
  __syncthreads();
  int f = threadIdx.x;
  float acc = bn[f];
  const float* wr = Wn + (size_t)f * ONF_;
  #pragma unroll 4
  for (int i = 0; i < ONF_; ++i) acc = fmaf(row[i], wr[i], acc);
  nf[(size_t)r * NFD + f] = acc;
}

// ef[r,e] = bf16(edge_fea[r,:41] . We[e,:41] + be[e])
__global__ __launch_bounds__(64) void k_embed_edge(const float* __restrict__ x,
                                                   const float* __restrict__ We,
                                                   const float* __restrict__ be,
                                                   bf16_t* __restrict__ ef) {
  __shared__ float row[OEF_];
  int r = blockIdx.x;
  for (int t = threadIdx.x; t < OEF_; t += 64) row[t] = x[(size_t)r * OEF_ + t];
  __syncthreads();
  int e = threadIdx.x;
  float acc = be[e];
  const float* wr = We + (size_t)e * OEF_;
  #pragma unroll
  for (int i = 0; i < OEF_; ++i) acc = fmaf(row[i], wr[i], acc);
  ef[(size_t)r * EFD + e] = f2bf(acc);
}

// ------------------------------ conv layer ---------------------------------
// One wave owns one (b,n): its 16 neighbor rows form exactly one 16-row WMMA
// A tile (16 x 320 bf16 staged in LDS). K-slabs of W (256x32 bf16) are staged
// cooperatively in LDS (async-to-LDS path when available) and shared by the 4
// waves of the block: W is read from global exactly once per block.
__global__ __launch_bounds__(128) void k_conv(const float* __restrict__ nf_in,
                                              const bf16_t* __restrict__ ef,
                                              const int* __restrict__ idx,
                                              const bf16_t* __restrict__ Wbf,   // [256,320] bf16
                                              const float* __restrict__ bvec,   // [256]
                                              const float* __restrict__ alphap, // [1]
                                              float* __restrict__ nf_out) {
  __shared__ bf16_t Asm[4][MM][KPAD];   // ~41 KB, padded rows
  __shared__ bf16_t Wsm[256][WSP];      // 20 KB K-slab, padded rows
  const int wav  = threadIdx.x >> 5;
  const int lane = threadIdx.x & 31;
  const int col  = lane & 15;
  const int hi   = lane >> 4;
  const int p    = blockIdx.x * 4 + wav;      // (b,n) pair id, exact cover
  const int bb   = p >> 8;                    // p / NN

  const float* selfp = nf_in + (size_t)p * NFD;
  const int*   idxp  = idx   + (size_t)p * MM;

  // Stage A = [self(128) | gathered(128) | edge(64)] per neighbor row.
  for (int m = 0; m < MM; ++m) {
    int j  = idxp[m];
    int jc = j < 0 ? 0 : j;                              // maximum(idx, 0)
    const float*  gp = nf_in + ((size_t)bb * NN + jc) * NFD;
    const bf16_t* ep = ef + ((size_t)p * MM + m) * EFD;
    bf16_t* dst = &Asm[wav][m][0];
    for (int t = lane; t < NFD; t += 32) dst[t]           = f2bf(selfp[t]);
    for (int t = lane; t < NFD; t += 32) dst[NFD + t]     = f2bf(gp[t]);
    for (int t = lane; t < EFD; t += 32) dst[2 * NFD + t] = ep[t];
  }

  v8f acc[16] = {};
  const bf16_t* arow = &Asm[wav][col][0];     // A: lane-half selects K subrange

  for (int kb = 0; kb < KTOT; kb += 32) {
    // Cooperative slab load: W[:, kb:kb+32] -> Wsm. 1024 16B chunks / 128 thr.
#if ASYNC_LDS
    #pragma unroll
    for (int i = 0; i < 8; ++i) {
      int c = i * 128 + threadIdx.x;
      int o = c >> 2;
      int j = (c & 3) * 8;
      __builtin_amdgcn_global_load_async_to_lds_b128(
          (g4p)(const void*)(Wbf + (size_t)o * KTOT + kb + j),
          (l4p)(void*)&Wsm[o][j], 0, 0);
    }
    asm volatile("s_wait_asynccnt 0" ::: "memory");
#else
    #pragma unroll
    for (int i = 0; i < 8; ++i) {
      int c = i * 128 + threadIdx.x;
      int o = c >> 2;
      int j = (c & 3) * 8;
      *(uint4*)&Wsm[o][j] = *(const uint4*)(Wbf + (size_t)o * KTOT + kb + j);
    }
#endif
    __syncthreads();
    v16bf a = ld16bf(arow + kb + hi * 8, arow + kb + 16 + hi * 8);
    #pragma unroll
    for (int c = 0; c < 16; ++c) {
      const bf16_t* wr = &Wsm[c * 16 + col][hi * 8];  // B: lane=N, K in two 8-runs (table order)
      v16bf bfr = ld16bf(wr, wr + 16);
      acc[c] = __builtin_amdgcn_wmma_f32_16x16x32_bf16(
          false, a, false, bfr, (short)0, acc[c], false, false);
    }
    __syncthreads();
  }

  // Epilogue: filter = sigmoid(gated[:,0:128]) * mask, core = softplus(gated[:,128:256])
  // sum over m: 8 rows per lane-half + cross-half shuffle.
  int idxv[8];
  #pragma unroll
  for (int v = 0; v < 8; ++v) idxv[v] = idxp[v + 8 * hi];
  const float alpha = alphap[0];
  float sums[8];
  #pragma unroll
  for (int c = 0; c < 8; ++c) {
    float bF = bvec[c * 16 + col];
    float bC = bvec[NFD + c * 16 + col];
    float s = 0.0f;
    #pragma unroll
    for (int v = 0; v < 8; ++v) {
      float fgate = sigmoidf_(acc[c][v] + bF);
      fgate = (idxv[v] >= 0) ? fgate : 0.0f;
      float core = softplusf_(acc[c + 8][v] + bC);
      s += fgate * core;
    }
    s += __shfl_xor(s, 16, 32);
    sums[c] = s;
  }
  if (hi == 0) {
    float* op = nf_out + (size_t)p * NFD;
    #pragma unroll
    for (int c = 0; c < 8; ++c) {
      int ch = c * 16 + col;
      op[ch] = softplusf_(alpha * selfp[ch] + sums[c]);
    }
  }
}

// --------------------------- final projection ------------------------------
// out[r, 0:64] = nf[r,:128] . Wf[f,:128] + bf; also store bf16 transposed
// copy nffT[b, f, n] for the WMMA DA-mixing GEMM.
__global__ __launch_bounds__(64) void k_final(const float* __restrict__ nf,
                                              const float* __restrict__ Wf,
                                              const float* __restrict__ bfv,
                                              float* __restrict__ out,
                                              bf16_t* __restrict__ nffT) {
  __shared__ float row[NFD];
  int r = blockIdx.x;
  for (int t = threadIdx.x; t < NFD; t += 64) row[t] = nf[(size_t)r * NFD + t];
  __syncthreads();
  int f = threadIdx.x;
  float acc = bfv[f];
  const float* wr = Wf + (size_t)f * NFD;
  #pragma unroll 4
  for (int i = 0; i < NFD; ++i) acc = fmaf(row[i], wr[i], acc);
  out[(size_t)r * 128 + f] = acc;
  int b = r >> 8, n = r & 255;
  nffT[((size_t)b * FH + f) * NN + n] = f2bf(acc);
}

// ------------------------- DA mixing (bf16 WMMA) ---------------------------
// out[b, i, 64+f] = sum_j DA[i,j] * nff[b,j,f]; per wave one 16x16 tile,
// K=256 in 8 wmma steps. nffT is [b][f][j] so B fragments are contiguous.
__global__ __launch_bounds__(128) void k_node1(const bf16_t* __restrict__ da,   // [256,256]
                                               const bf16_t* __restrict__ nffT, // [B,64,256]
                                               float* __restrict__ out) {
  const int wav  = threadIdx.x >> 5;
  const int lane = threadIdx.x & 31;
  const int col  = lane & 15;
  const int hi   = lane >> 4;
  int t   = blockIdx.x * 4 + wav;   // 4096 tiles, exact cover
  int b   = t >> 6;
  int rem = t & 63;
  int i0  = (rem >> 2) * 16;
  int f0  = (rem & 3) * 16;
  v8f acc = {};
  const bf16_t* arow = da   + (size_t)(i0 + col) * NN;
  const bf16_t* brow = nffT + ((size_t)b * FH + f0 + col) * NN;
  #pragma unroll
  for (int kb = 0; kb < NN; kb += 32) {
    v16bf a  = ld16bf(arow + kb + hi * 8,  arow + kb + 16 + hi * 8);
    v16bf bf = ld16bf(brow + kb + hi * 8,  brow + kb + 16 + hi * 8);
    acc = __builtin_amdgcn_wmma_f32_16x16x32_bf16(
        false, a, false, bf, (short)0, acc, false, false);
  }
  #pragma unroll
  for (int v = 0; v < 8; ++v) {
    int i = i0 + v + 8 * hi;
    out[((size_t)b * NN + i) * 128 + 64 + f0 + col] = acc[v];
  }
}

// ------------------------------- launcher ----------------------------------

extern "C" void kernel_launch(void* const* d_in, const int* in_sizes, int n_in,
                              void* d_out, int out_size, void* d_ws, size_t ws_size,
                              hipStream_t stream) {
  (void)in_sizes; (void)n_in; (void)out_size; (void)ws_size;
  const float* node_fea = (const float*)d_in[0];
  const float* edge_fea = (const float*)d_in[1];
  const int*   eidx     = (const int*)d_in[2];
  const float* dis      = (const float*)d_in[3];
  const float* Wn  = (const float*)d_in[4];
  const float* bn  = (const float*)d_in[5];
  const float* We  = (const float*)d_in[6];
  const float* be  = (const float*)d_in[7];
  const float* W1  = (const float*)d_in[8];
  const float* b1  = (const float*)d_in[9];
  const float* a1  = (const float*)d_in[10];
  const float* W2  = (const float*)d_in[11];
  const float* b2  = (const float*)d_in[12];
  const float* a2  = (const float*)d_in[13];
  const float* W3  = (const float*)d_in[14];
  const float* b3  = (const float*)d_in[15];
  const float* a3  = (const float*)d_in[16];
  const float* Wf  = (const float*)d_in[17];
  const float* bfv = (const float*)d_in[18];
  const float* DAw = (const float*)d_in[19];
  const float* DAb = (const float*)d_in[20];
  float* out = (float*)d_out;

  char* ws = (char*)d_ws;
  float*  nf0  = (float*)(ws);
  float*  nf1  = (float*)(ws + ((size_t)8  << 20));
  bf16_t* efb  = (bf16_t*)(ws + ((size_t)16 << 20));
  bf16_t* wb1  = (bf16_t*)(ws + ((size_t)48 << 20));
  bf16_t* wb2  = wb1 + 81920;           // 256*320
  bf16_t* wb3  = wb2 + 81920;
  bf16_t* dab  = wb3 + 81920;           // 256*256
  bf16_t* nffT = dab + 65536;           // 64*64*256

  k_cvt_bf16<<<320, 256, 0, stream>>>(W1, wb1, 81920);
  k_cvt_bf16<<<320, 256, 0, stream>>>(W2, wb2, 81920);
  k_cvt_bf16<<<320, 256, 0, stream>>>(W3, wb3, 81920);
  k_da<<<256, 256, 0, stream>>>(dis, DAw, DAb, dab);

  k_embed_node<<<BB * NN, 128, 0, stream>>>(node_fea, Wn, bn, nf0);
  k_embed_edge<<<BB * NN * MM, 64, 0, stream>>>(edge_fea, We, be, efb);

  k_conv<<<BB * NN / 4, 128, 0, stream>>>(nf0, efb, eidx, wb1, b1, a1, nf1);
  k_conv<<<BB * NN / 4, 128, 0, stream>>>(nf1, efb, eidx, wb2, b2, a2, nf0);
  k_conv<<<BB * NN / 4, 128, 0, stream>>>(nf0, efb, eidx, wb3, b3, a3, nf1);

  k_final<<<BB * NN, 64, 0, stream>>>(nf1, Wf, bfv, out, nffT);
  k_node1<<<BB * NN / 4 / 4, 128, 0, stream>>>(dab, nffT, out);
}
